// GraphClassifier_30124900614232
// MI455X (gfx1250) — matrix operand order repaired and checked by reading, not compiled, not run on hardware
//
#include <hip/hip_runtime.h>
#include <hip/hip_bf16.h>

// ---------------------------------------------------------------------------
// Types for CDNA5 WMMA
// ---------------------------------------------------------------------------
typedef __attribute__((ext_vector_type(16))) __bf16 bf16x16;
typedef __attribute__((ext_vector_type(8)))  __bf16 bf16x8;
typedef __attribute__((ext_vector_type(8)))  float  v8f;

#define H_DIM 128
#define LDS_STRIDE 136  // 128 + 8 pad (272B rows -> conflict-free b128 LDS loads)

// ---------------------------------------------------------------------------
// Small utility kernels
// ---------------------------------------------------------------------------
__global__ void zero_f32_kernel(float* __restrict__ p, long n) {
    long i = (long)blockIdx.x * blockDim.x + threadIdx.x;
    long stride = (long)gridDim.x * blockDim.x;
    for (; i < n; i += stride) p[i] = 0.0f;
}

__global__ void degree_kernel(const int* __restrict__ dst, float* __restrict__ deg, int E) {
    int e = blockIdx.x * blockDim.x + threadIdx.x;
    if (e < E) atomicAdd(&deg[dst[e]], 1.0f);
}

// dinv[i] = rsqrt(indeg[i] + 1)   (self-loop included; always > 0)
__global__ void dinv_kernel(float* __restrict__ deg_dinv, int N) {
    int i = blockIdx.x * blockDim.x + threadIdx.x;
    if (i < N) deg_dinv[i] = rsqrtf(deg_dinv[i] + 1.0f);
}

// ---------------------------------------------------------------------------
// GEMM: out[i,:] = dinv[i] * (X[i,:] @ W)   with bf16x3 split-precision WMMA
//   X: [N,128] f32 row-major, W: [128,128] f32 row-major (W[k][n] = W[k*128+n])
//   block = 256 threads (8 waves); wave w computes rows [blk*128 + w*16, +16)
// ---------------------------------------------------------------------------
__device__ inline bf16x16 lds_load16(const __bf16* p) {
    bf16x8 a = *(const bf16x8*)p;        // 16B aligned
    bf16x8 b = *(const bf16x8*)(p + 16);
    return __builtin_shufflevector(a, b, 0,1,2,3,4,5,6,7,8,9,10,11,12,13,14,15);
}

__global__ __launch_bounds__(256) void gcn_gemm_kernel(
    const float* __restrict__ X, const float* __restrict__ W,
    const float* __restrict__ dinv, float* __restrict__ out, int N)
{
    __shared__ __bf16 WhiT[H_DIM * LDS_STRIDE];   // transposed: WhiT[n][k]
    __shared__ __bf16 WloT[H_DIM * LDS_STRIDE];

    const int tid = threadIdx.x;
    // Stage W: split into bf16 hi/lo, transposed so B fragments are contiguous.
    for (int i = tid; i < H_DIM * H_DIM; i += 256) {
        const int k = i >> 7, n = i & 127;
        const float w  = W[i];                 // W[k][n]
        const __bf16 hi = (__bf16)w;
        const __bf16 lo = (__bf16)(w - (float)hi);
        WhiT[n * LDS_STRIDE + k] = hi;
        WloT[n * LDS_STRIDE + k] = lo;
    }
    __syncthreads();

    const int wave    = tid >> 5;
    const int lane    = tid & 31;
    const int laneHi  = lane >> 4;     // 0: lanes 0-15, 1: lanes 16-31
    const int lane16  = lane & 15;
    const int rowBase = blockIdx.x * 128 + wave * 16;
    const int rowA    = min(rowBase + lane16, N - 1);   // clamp OOB loads

    v8f acc[8];
    #pragma unroll
    for (int j = 0; j < 8; ++j) acc[j] = (v8f)0.0f;

    const float* xrow = X + (size_t)rowA * H_DIM;

    #pragma unroll
    for (int k0 = 0; k0 < H_DIM; k0 += 32) {
        // A fragment (16-bit A layout): elem i   <- X[row][k0 + 8*laneHi + i]
        //                               elem 8+i <- X[row][k0 + 16 + 8*laneHi + i]
        const float4 c0 = *(const float4*)(xrow + k0 + 8 * laneHi);
        const float4 c1 = *(const float4*)(xrow + k0 + 8 * laneHi + 4);
        const float4 c2 = *(const float4*)(xrow + k0 + 16 + 8 * laneHi);
        const float4 c3 = *(const float4*)(xrow + k0 + 16 + 8 * laneHi + 4);
        float af[16] = {c0.x,c0.y,c0.z,c0.w, c1.x,c1.y,c1.z,c1.w,
                        c2.x,c2.y,c2.z,c2.w, c3.x,c3.y,c3.z,c3.w};
        bf16x16 ahi, alo;
        #pragma unroll
        for (int i = 0; i < 16; ++i) {
            const __bf16 h = (__bf16)af[i];
            ahi[i] = h;
            alo[i] = (__bf16)(af[i] - (float)h);
        }

        const int kb = k0 + 8 * laneHi;   // B fragment k-base for this lane
        #pragma unroll
        for (int j = 0; j < 8; ++j) {
            const int boff = (j * 16 + lane16) * LDS_STRIDE + kb;
            const bf16x16 bhi = lds_load16(&WhiT[boff]);
            const bf16x16 blo = lds_load16(&WloT[boff]);
            // bf16x3: hi*hi + lo*hi + hi*lo  (f32 accumulate)
            acc[j] = __builtin_amdgcn_wmma_f32_16x16x32_bf16(
                        false, ahi, false, bhi, (short)0, acc[j], false, false);
            acc[j] = __builtin_amdgcn_wmma_f32_16x16x32_bf16(
                        false, alo, false, bhi, (short)0, acc[j], false, false);
            acc[j] = __builtin_amdgcn_wmma_f32_16x16x32_bf16(
                        false, ahi, false, blo, (short)0, acc[j], false, false);
        }
    }

    // C/D layout: VGPR r holds M = r + 8*laneHi, Ncol = lane16 (+16*j).
    #pragma unroll
    for (int r = 0; r < 8; ++r) {
        const int row = rowBase + r + 8 * laneHi;
        if (row < N) {
            const float dv = dinv[row];
            float* orow = out + (size_t)row * H_DIM + lane16;
            #pragma unroll
            for (int j = 0; j < 8; ++j) orow[j * 16] = acc[j][r] * dv;
        }
    }
}

// ---------------------------------------------------------------------------
// Edge scatter: one wave per edge; lane handles 4 consecutive features.
//   accum[dst] += hs[src]       (hs already scaled by dinv[src])
// ---------------------------------------------------------------------------
__global__ __launch_bounds__(256) void scatter_kernel(
    const int* __restrict__ src, const int* __restrict__ dst,
    const float* __restrict__ hs, float* __restrict__ accum, int E)
{
    const long idx = (long)blockIdx.x * blockDim.x + threadIdx.x;
    if (idx >= (long)E * 32) return;
    const int e    = (int)(idx >> 5);
    const int lane = (int)(idx & 31);
    const int s = src[e];
    const int d = dst[e];
    const float4 v = *(const float4*)(hs + (size_t)s * H_DIM + lane * 4);
    float* o = accum + (size_t)d * H_DIM + lane * 4;
    atomicAdd(o + 0, v.x);
    atomicAdd(o + 1, v.y);
    atomicAdd(o + 2, v.z);
    atomicAdd(o + 3, v.w);
}

// out[i,f] = relu( dinv[i] * (accum[i,f] + hs[i,f]) + b[f] ), in place on accum
__global__ void finalize_kernel(float* __restrict__ accum,
                                const float* __restrict__ hs,
                                const float* __restrict__ dinv,
                                const float* __restrict__ b, int N)
{
    const long t = (long)blockIdx.x * blockDim.x + threadIdx.x;
    if (t >= (long)N * H_DIM) return;
    const int i = (int)(t >> 7);
    const int f = (int)(t & 127);
    const float y = dinv[i] * (accum[t] + hs[t]) + b[f];
    accum[t] = fmaxf(y, 0.0f);
}

// ---------------------------------------------------------------------------
// Global mean pool pieces
// ---------------------------------------------------------------------------
__global__ void pool_kernel(const float* __restrict__ h,
                            const int* __restrict__ batch,
                            float* __restrict__ sums, float* __restrict__ cnt, int N)
{
    const long t = (long)blockIdx.x * blockDim.x + threadIdx.x;
    if (t >= (long)N * H_DIM) return;
    const int i = (int)(t >> 7);
    const int f = (int)(t & 127);
    const int g = batch[i];
    atomicAdd(&sums[g * H_DIM + f], h[t]);
    if (f == 0) atomicAdd(&cnt[g], 1.0f);
}

// One block (128 threads) per graph: mean -> fc1 relu -> fc2 -> log_softmax
__global__ __launch_bounds__(128) void head_kernel(
    const float* __restrict__ sums, const float* __restrict__ cnt,
    const float* __restrict__ fcW1, const float* __restrict__ fcb1,
    const float* __restrict__ fcW2, const float* __restrict__ fcb2,
    float* __restrict__ out)
{
    __shared__ float m[H_DIM];
    __shared__ float z[H_DIM];
    __shared__ float l[10];
    __shared__ float red[2];
    const int b = blockIdx.x, t = threadIdx.x;

    const float c = fmaxf(cnt[b], 1.0f);
    m[t] = sums[b * H_DIM + t] / c;
    __syncthreads();

    float s = fcb1[t];
    for (int k = 0; k < H_DIM; ++k) s += m[k] * fcW1[k * H_DIM + t];
    z[t] = fmaxf(s, 0.0f);
    __syncthreads();

    if (t < 10) {
        float s2 = fcb2[t];
        for (int k = 0; k < H_DIM; ++k) s2 += z[k] * fcW2[k * 10 + t];
        l[t] = s2;
    }
    __syncthreads();

    if (t == 0) {
        float mx = l[0];
        for (int i = 1; i < 10; ++i) mx = fmaxf(mx, l[i]);
        float se = 0.0f;
        for (int i = 0; i < 10; ++i) se += __expf(l[i] - mx);
        red[0] = mx;
        red[1] = __logf(se);
    }
    __syncthreads();
    if (t < 10) out[b * 10 + t] = l[t] - red[0] - red[1];
}

// ---------------------------------------------------------------------------
// Host-side launcher
// ---------------------------------------------------------------------------
extern "C" void kernel_launch(void* const* d_in, const int* in_sizes, int n_in,
                              void* d_out, int out_size, void* d_ws, size_t ws_size,
                              hipStream_t stream) {
    const float* x     = (const float*)d_in[0];
    const int*   ei    = (const int*)  d_in[1];
    const int*   batch = (const int*)  d_in[2];
    const float* W1    = (const float*)d_in[3];
    const float* b1    = (const float*)d_in[4];
    const float* W2    = (const float*)d_in[5];
    const float* b2    = (const float*)d_in[6];
    const float* fcW1  = (const float*)d_in[7];
    const float* fcb1  = (const float*)d_in[8];
    const float* fcW2  = (const float*)d_in[9];
    const float* fcb2  = (const float*)d_in[10];

    const int N = in_sizes[0] / H_DIM;
    const int E = in_sizes[1] / 2;
    const int* src = ei;         // edge_index[0]
    const int* dst = ei + E;     // edge_index[1]

    // Workspace carve-up (all float):
    //   dinv [Npad] | buf0 [N*128] (hs) | buf1 [N*128] (accum/y) | sums [64*128] | cnt [64]
    float* ws   = (float*)d_ws;
    const size_t Npad = ((size_t)N + 3) & ~(size_t)3;
    float* dinv = ws;
    float* buf0 = dinv + Npad;
    float* buf1 = buf0 + (size_t)N * H_DIM;
    float* sums = buf1 + (size_t)N * H_DIM;
    float* cnt  = sums + 64 * H_DIM;

    const long NH = (long)N * H_DIM;
    const int  TB = 256;
    const int  nBlkNH   = (int)((NH + TB - 1) / TB);
    const int  nBlkN    = (N + TB - 1) / TB;
    const int  nBlkE    = (E + TB - 1) / TB;
    const int  nBlkScat = (int)(((long)E * 32 + TB - 1) / TB);
    const int  nBlkGemm = (N + 127) / 128;

    // Degree + symmetric normalization factor
    zero_f32_kernel<<<nBlkN, TB, 0, stream>>>(dinv, N);
    degree_kernel<<<nBlkE, TB, 0, stream>>>(dst, dinv, E);
    dinv_kernel<<<nBlkN, TB, 0, stream>>>(dinv, N);

    // ---- Layer 1: hs = dinv * (x @ W1); accum = scatter; y = relu(dinv*(accum+hs)+b1)
    gcn_gemm_kernel<<<nBlkGemm, TB, 0, stream>>>(x, W1, dinv, buf0, N);
    zero_f32_kernel<<<nBlkNH, TB, 0, stream>>>(buf1, NH);
    scatter_kernel<<<nBlkScat, TB, 0, stream>>>(src, dst, buf0, buf1, E);
    finalize_kernel<<<nBlkNH, TB, 0, stream>>>(buf1, buf0, dinv, b1, N);

    // ---- Layer 2 (input = buf1)
    gcn_gemm_kernel<<<nBlkGemm, TB, 0, stream>>>(buf1, W2, dinv, buf0, N);
    zero_f32_kernel<<<nBlkNH, TB, 0, stream>>>(buf1, NH);
    scatter_kernel<<<nBlkScat, TB, 0, stream>>>(src, dst, buf0, buf1, E);
    finalize_kernel<<<nBlkNH, TB, 0, stream>>>(buf1, buf0, dinv, b2, N);

    // ---- Pool + classification head
    zero_f32_kernel<<<((64 * H_DIM + 64) + TB - 1) / TB, TB, 0, stream>>>(sums, 64 * H_DIM + 64);
    pool_kernel<<<nBlkNH, TB, 0, stream>>>(buf1, batch, sums, cnt, N);
    head_kernel<<<64, H_DIM, 0, stream>>>(sums, cnt, fcW1, fcb1, fcW2, fcb2, (float*)d_out);
}
